// GraphSAGE_14276471292553
// MI455X (gfx1250) — matrix operand order repaired and checked by reading, not compiled, not run on hardware
//
#include <hip/hip_runtime.h>
#include <hip/hip_bf16.h>

// ---------------------------------------------------------------------------
// CDNA5 (gfx1250) types + feature detection
// ---------------------------------------------------------------------------
typedef __attribute__((ext_vector_type(16))) __bf16 v16bf;
typedef __attribute__((ext_vector_type(8)))  float  v8f;
typedef __attribute__((ext_vector_type(4)))  int    v4i;

union Frag16 {
    v16bf v;
    uint4 q[2];
};

#if defined(__has_builtin)
#  if __has_builtin(__builtin_amdgcn_global_load_async_to_lds_b128)
#    define USE_ASYNC_COPY 1
#  endif
#  if __has_builtin(__builtin_amdgcn_s_wait_asynccnt)
#    define HAVE_WAIT_ASYNC_BUILTIN 1
#  endif
#endif
#ifndef USE_ASYNC_COPY
#  define USE_ASYNC_COPY 0
#endif
#ifndef HAVE_WAIT_ASYNC_BUILTIN
#  define HAVE_WAIT_ASYNC_BUILTIN 0
#endif

// f32 -> bf16 (round-to-nearest-even) as raw u16
__device__ __forceinline__ unsigned short f2bf_u16(float f) {
    unsigned u = __builtin_bit_cast(unsigned, f);
    return (unsigned short)((u + 0x7FFFu + ((u >> 16) & 1u)) >> 16);
}
__device__ __forceinline__ float bf2f(unsigned short h) {
    return __builtin_bit_cast(float, (unsigned)h << 16);
}

// 16-byte global -> LDS copy; async (ASYNCcnt-tracked, bypasses VGPRs) when
// the gfx1250 builtin is available, plain copy otherwise.
__device__ __forceinline__ void lds_copy16(unsigned short* ldst,
                                           const unsigned short* gsrc) {
#if USE_ASYNC_COPY
    __builtin_amdgcn_global_load_async_to_lds_b128(
        (__attribute__((address_space(1))) v4i*)(unsigned short*)gsrc,
        (__attribute__((address_space(3))) v4i*)ldst,
        0, 0);
#else
    *(uint4*)ldst = *(const uint4*)gsrc;
#endif
}

__device__ __forceinline__ void wait_async_copies() {
#if USE_ASYNC_COPY
#  if HAVE_WAIT_ASYNC_BUILTIN
    __builtin_amdgcn_s_wait_asynccnt(0);
#  else
    asm volatile("s_wait_asynccnt 0" ::: "memory");
#  endif
#endif
}

// ---------------------------------------------------------------------------
// Utility kernels
// ---------------------------------------------------------------------------
__global__ void zero_f32(float* p, long long n) {
    long long i = (long long)blockIdx.x * blockDim.x + threadIdx.x;
    long long stride = (long long)gridDim.x * blockDim.x;
    for (; i < n; i += stride) p[i] = 0.0f;
}

__global__ void f32_to_bf16(const float* __restrict__ in,
                            unsigned short* __restrict__ out, long long n) {
    long long i = (long long)blockIdx.x * blockDim.x + threadIdx.x;
    long long stride = (long long)gridDim.x * blockDim.x;
    for (; i < n; i += stride) out[i] = f2bf_u16(in[i]);
}

__global__ void degree_count(const int* __restrict__ dst, float* cnt, int E) {
    int e = blockIdx.x * blockDim.x + threadIdx.x;
    if (e < E) atomicAdd(&cnt[dst[e]], 1.0f);
}

// Gather f32 X[src] rows (float4) and scatter-add into AGG[dst].
// One thread = 4 consecutive floats of one edge. perEdgeShift = log2(D/4).
__global__ void edge_scatter_add_f32(const float* __restrict__ X,
                                     const int* __restrict__ src,
                                     const int* __restrict__ dst,
                                     float* AGG, long long total,
                                     int D, int perEdgeShift) {
    long long t = (long long)blockIdx.x * blockDim.x + threadIdx.x;
    if (t >= total) return;
    int e = (int)(t >> perEdgeShift);
    int c = (int)(t & ((1 << perEdgeShift) - 1)) << 2;
    int s = src[e], d = dst[e];
    float4 v = *(const float4*)(X + (size_t)s * D + c);
    float* out = AGG + (size_t)d * D + c;
    atomicAdd(out + 0, v.x);
    atomicAdd(out + 1, v.y);
    atomicAdd(out + 2, v.z);
    atomicAdd(out + 3, v.w);
}

// Gather bf16 H[src] rows (8 bf16 = 16B per thread) and scatter-add f32.
// perEdgeShift = log2(D/8).
__global__ void edge_scatter_add_bf16(const unsigned short* __restrict__ H,
                                      const int* __restrict__ src,
                                      const int* __restrict__ dst,
                                      float* AGG, long long total,
                                      int D, int perEdgeShift) {
    long long t = (long long)blockIdx.x * blockDim.x + threadIdx.x;
    if (t >= total) return;
    int e = (int)(t >> perEdgeShift);
    int c = (int)(t & ((1 << perEdgeShift) - 1)) << 3;
    int s = src[e], d = dst[e];
    uint4 q = *(const uint4*)(H + (size_t)s * D + c);
    unsigned words[4] = {q.x, q.y, q.z, q.w};
    float* out = AGG + (size_t)d * D + c;
    #pragma unroll
    for (int j = 0; j < 4; ++j) {
        atomicAdd(out + 2 * j + 0, bf2f((unsigned short)(words[j] & 0xFFFFu)));
        atomicAdd(out + 2 * j + 1, bf2f((unsigned short)(words[j] >> 16)));
    }
}

// ---------------------------------------------------------------------------
// Fused SAGE GEMM (bf16 WMMA, fp32 accumulate):
//   C = relu?( pair0 @ B1^T + pair1 @ B2^T + bias )
// pair0 A-source: either f32 A1f with mean divide by max(cnt,1), or bf16 A1b.
// pair1 A-source: bf16 A2b (null -> single pair).
// B1/B2: bf16 [Nout,K] row-major (W layout -> computes A @ W^T).
// Block 256 threads = 8 waves; tile 128(M) x 64(N); K-step 32.
// bf16 tiles go global->LDS via GLOBAL_LOAD_ASYNC_TO_LDS_B128 when available.
// ---------------------------------------------------------------------------
#define TM 128
#define TN 64
#define TK 32
#define LDT 40   // padded LDS row stride (bf16 units): 80B rows

__global__ void __launch_bounds__(256)
sage_wmma_gemm(const float* A1f, const float* cnt,
               const unsigned short* A1b,
               const unsigned short* A2b,
               const unsigned short* __restrict__ B1,
               const unsigned short* __restrict__ B2,
               const float* __restrict__ bias,
               void* Cout, int outBf16,
               int M, int Nout, int K, int relu) {
    __shared__ unsigned short As[TM * LDT];   // 10 KB
    __shared__ unsigned short Bs[TN * LDT];   // 5 KB

    const int tid  = threadIdx.x;
    const int m0   = blockIdx.x * TM;
    const int n0   = blockIdx.y * TN;
    const int w    = tid >> 5;          // wave id 0..7
    const int lane = tid & 31;
    const int grp  = lane >> 4;         // lane group (ISA 16-bit layouts)
    const int l16  = lane & 15;

    const int KT = A2b ? (2 * K) : K;

    v8f acc[4];
    acc[0] = v8f{}; acc[1] = v8f{}; acc[2] = v8f{}; acc[3] = v8f{};

    for (int kk = 0; kk < KT; kk += TK) {
        const int pair = (kk >= K) ? 1 : 0;
        const int kb   = kk - (pair ? K : 0);

        // ---- stage A tile (TM x TK bf16) ----
        if (pair == 0 && A1f) {
            // mean operand: f32 load, divide by degree, convert
            for (int i = tid; i < TM * TK; i += 256) {
                int r = i >> 5, k = i & 31;
                int gr = m0 + r;
                float v = 0.0f;
                if (gr < M) {
                    v = A1f[(size_t)gr * K + kb + k];
                    if (cnt) v *= (1.0f / fmaxf(cnt[gr], 1.0f));
                }
                As[r * LDT + k] = f2bf_u16(v);
            }
        } else {
            const unsigned short* Asrc = pair ? A2b : A1b;
            // 4 x 16B chunks per 32-elem row
            for (int i = tid; i < TM * 4; i += 256) {
                int r = i >> 2, ch = i & 3;
                int gr = m0 + r;
                unsigned short* ldst = &As[r * LDT + ch * 8];
                if (gr < M)
                    lds_copy16(ldst, Asrc + (size_t)gr * K + kb + ch * 8);
                else
                    *(uint4*)ldst = uint4{0u, 0u, 0u, 0u};
            }
        }
        // ---- stage B tile (TN x TK bf16): rows are output columns ----
        {
            const unsigned short* Bsrc = pair ? B2 : B1;
            for (int i = tid; i < TN * 4; i += 256) {
                int r = i >> 2, ch = i & 3;
                lds_copy16(&Bs[r * LDT + ch * 8],
                           Bsrc + (size_t)(n0 + r) * K + kb + ch * 8);
            }
        }
        wait_async_copies();
        __syncthreads();

        // ---- fragments per ISA 7.12.2 lane layouts ----
        // A (16x32): lane l -> row l%16; halves 0..7 = K[grp*8..+7],
        //            halves 8..15 = K[16+grp*8..+7]
        Frag16 a;
        const unsigned short* ap = &As[(w * 16 + l16) * LDT];
        a.q[0] = *(const uint4*)(ap + grp * 8);
        a.q[1] = *(const uint4*)(ap + 16 + grp * 8);

        // B (32x16): lane l -> col l%16; halves 0..7 = K[grp*8..+7],
        //            halves 8..15 = K[16+grp*8..+7]
        #pragma unroll
        for (int t = 0; t < 4; ++t) {
            Frag16 b;
            const unsigned short* bp = &Bs[(t * 16 + l16) * LDT];
            b.q[0] = *(const uint4*)(bp + grp * 8);
            b.q[1] = *(const uint4*)(bp + 16 + grp * 8);
            acc[t] = __builtin_amdgcn_wmma_f32_16x16x32_bf16(
                false, a.v, false, b.v, (short)0, acc[t], false, false);
        }
        __syncthreads();
    }

    // ---- epilogue: C/D layout: vgpr r, lane l -> row r+(l/16)*8, col l%16
    #pragma unroll
    for (int t = 0; t < 4; ++t) {
        int col = n0 + t * 16 + l16;
        float bv = bias ? bias[col] : 0.0f;
        #pragma unroll
        for (int r = 0; r < 8; ++r) {
            int row = m0 + w * 16 + r + grp * 8;
            if (row < M) {
                float v = acc[t][r] + bv;
                if (relu) v = fmaxf(v, 0.0f);
                if (outBf16)
                    ((unsigned short*)Cout)[(size_t)row * Nout + col] = f2bf_u16(v);
                else
                    ((float*)Cout)[(size_t)row * Nout + col] = v;
            }
        }
    }
}

// ---------------------------------------------------------------------------
// Host-side orchestration
// ---------------------------------------------------------------------------
extern "C" void kernel_launch(void* const* d_in, const int* in_sizes, int n_in,
                              void* d_out, int out_size, void* d_ws, size_t ws_size,
                              hipStream_t stream) {
    (void)n_in; (void)out_size; (void)ws_size;

    const float* x     = (const float*)d_in[0];
    const int*   eidx  = (const int*)  d_in[1];
    const float* w1_l  = (const float*)d_in[2];
    const float* w1_r  = (const float*)d_in[3];
    const float* b1    = (const float*)d_in[4];
    const float* w2_l  = (const float*)d_in[5];
    const float* w2_r  = (const float*)d_in[6];
    const float* b2    = (const float*)d_in[7];
    const float* w_lin = (const float*)d_in[8];
    const float* b_lin = (const float*)d_in[9];
    float* out = (float*)d_out;

    const int hidden = in_sizes[4];                 // 512
    const int dout   = in_sizes[9];                 // 128
    const int d_in_f = in_sizes[2] / hidden;        // 128
    const int N      = in_sizes[0] / d_in_f;        // 50000
    const int E      = in_sizes[1] / 2;             // 800000

    const int* src = eidx;
    const int* dst = eidx + E;

    // workspace layout
    float* cnt  = (float*)d_ws;                          // [N]
    float* agg1 = cnt + N;                               // [N, d_in]   f32
    float* agg2 = agg1 + (size_t)N * d_in_f;             // [N, hidden] f32
    unsigned short* xbf  = (unsigned short*)(agg2 + (size_t)N * hidden); // [N, d_in]
    unsigned short* h1bf = xbf + (size_t)N * d_in_f;     // [N, hidden]
    unsigned short* w1l_bf = h1bf + (size_t)N * hidden;
    unsigned short* w1r_bf = w1l_bf + in_sizes[2];
    unsigned short* w2l_bf = w1r_bf + in_sizes[3];
    unsigned short* w2r_bf = w2l_bf + in_sizes[5];
    unsigned short* wlin_bf = w2r_bf + in_sizes[6];

    // 1) zero atomic accumulators (every call)
    long long nz = (long long)N * (1 + d_in_f + hidden);
    zero_f32<<<2048, 256, 0, stream>>>(cnt, nz);

    // 2) bf16 copies of x and weights (one shot per call)
    f32_to_bf16<<<2048, 256, 0, stream>>>(x, xbf, (long long)N * d_in_f);
    f32_to_bf16<<<256, 256, 0, stream>>>(w1_l, w1l_bf, in_sizes[2]);
    f32_to_bf16<<<256, 256, 0, stream>>>(w1_r, w1r_bf, in_sizes[3]);
    f32_to_bf16<<<256, 256, 0, stream>>>(w2_l, w2l_bf, in_sizes[5]);
    f32_to_bf16<<<256, 256, 0, stream>>>(w2_r, w2r_bf, in_sizes[6]);
    f32_to_bf16<<<256, 256, 0, stream>>>(w_lin, wlin_bf, in_sizes[8]);

    // 3) degree counts
    degree_count<<<(E + 255) / 256, 256, 0, stream>>>(dst, cnt, E);

    // 4) scatter x -> agg1 (f32 gather, f32 atomics)
    {
        int shift = 31 - __builtin_clz((unsigned)(d_in_f >> 2)); // log2(D/4)
        long long total = (long long)E << shift;
        edge_scatter_add_f32<<<(unsigned)((total + 255) / 256), 256, 0, stream>>>(
            x, src, dst, agg1, total, d_in_f, shift);
    }

    // 5) layer 1: h1 = relu(mean(agg1) @ w1_l^T + x @ w1_r^T + b1) -> bf16
    {
        dim3 grid((N + TM - 1) / TM, hidden / TN);
        sage_wmma_gemm<<<grid, 256, 0, stream>>>(
            agg1, cnt, nullptr, xbf, w1l_bf, w1r_bf, b1,
            h1bf, 1, N, hidden, d_in_f, 1);
    }

    // 6) scatter h1 (bf16 gather) -> agg2 (f32 atomics)
    {
        int shift = 31 - __builtin_clz((unsigned)(hidden >> 3)); // log2(D/8)
        long long total = (long long)E << shift;
        edge_scatter_add_bf16<<<(unsigned)((total + 255) / 256), 256, 0, stream>>>(
            h1bf, src, dst, agg2, total, hidden, shift);
    }

    // 7) layer 2, in-place into h1bf (each block writes only rows it reads)
    {
        dim3 grid((N + TM - 1) / TM, hidden / TN);
        sage_wmma_gemm<<<grid, 256, 0, stream>>>(
            agg2, cnt, nullptr, h1bf, w2l_bf, w2r_bf, b2,
            h1bf, 1, N, hidden, hidden, 1);
    }

    // 8) head: out = h2 @ w_lin^T + b_lin (f32 out)
    {
        dim3 grid((N + TM - 1) / TM, dout / TN);
        sage_wmma_gemm<<<grid, 256, 0, stream>>>(
            nullptr, nullptr, h1bf, nullptr, wlin_bf, nullptr, b_lin,
            out, 0, N, dout, hidden, 0);
    }
}
